// CausalSelfAttention_50259707298340
// MI455X (gfx1250) — matrix operand order repaired and checked by reading, not compiled, not run on hardware
//
#include <hip/hip_runtime.h>
#include <cmath>
#include <cstdint>

typedef __attribute__((ext_vector_type(16))) _Float16 v16h;
typedef __attribute__((ext_vector_type(8)))  _Float16 v8h;
typedef __attribute__((ext_vector_type(8)))  float    v8f;
typedef __attribute__((ext_vector_type(4)))  float    v4f;

#define WPB 4  // waves per block (wave32)

// Same-wave cross-lane LDS visibility: DS ops from this wave complete in
// order; wait for them and stop the scheduler from hoisting loads above.
__device__ __forceinline__ void wave_lds_fence() {
  __builtin_amdgcn_wave_barrier();
  asm volatile("s_wait_dscnt 0" ::: "memory");
  __builtin_amdgcn_wave_barrier();
}

__device__ __forceinline__ void wait_async0() {
  __builtin_amdgcn_wave_barrier();
  asm volatile("s_wait_asynccnt 0" ::: "memory");
  __builtin_amdgcn_wave_barrier();
}

// Async copy, 16B/lane, global->LDS (ASYNCcnt). INST_OFFSET is applied to
// both the LDS and the global address, so one base register pair serves all
// eight 512B slices of the 4KB tile.
#define ASYNC_B128_OFF(lds, gaddr, off)                                  \
  asm volatile("global_load_async_to_lds_b128 %0, %1, off offset:" #off  \
               :: "v"(lds), "v"(gaddr) : "memory")

__device__ __forceinline__ v8f wmma16(v16h a, v16h b, v8f c) {
  // D = A(16x32 f16) * B(32x16 f16) + C(16x16 f32)
  return __builtin_amdgcn_wmma_f32_16x16x32_f16(
      /*neg_a=*/false, a, /*neg_b=*/false, b,
      /*c_mod=*/(short)0, c, /*reuse_a=*/false, /*reuse_b=*/false);
}

// A operand (16x32, rows m_off..m_off+15) from a row-major f16 32x32 tile.
// Lane l (l<16): row=l,    halves 0-7 = K 0..7,  halves 8-15 = K 16..23
// Lane l (>=16): row=l-16, halves 0-7 = K 8..15, halves 8-15 = K 24..31
__device__ __forceinline__ v16h lds_A(const _Float16* base, int lane, int m_off) {
  int row = (lane & 15) + m_off;
  int kb  = (lane >> 4) << 3;
  const _Float16* p = base + row * 32 + kb;
  v8h lo = *(const v8h*)(p);
  v8h hi = *(const v8h*)(p + 16);
  v16h r;
#pragma unroll
  for (int i = 0; i < 8; ++i) { r[i] = lo[i]; r[i + 8] = hi[i]; }
  return r;
}

// Same A layout, but source tile is row-major f32 (x kept f32 in LDS after
// the async copy); convert while assembling the operand.
__device__ __forceinline__ v16h lds_A_f32(const float* base, int lane, int m_off) {
  int row = (lane & 15) + m_off;
  int kb  = (lane >> 4) << 3;
  const float* p = base + row * 32 + kb;
  v4f a0 = *(const v4f*)(p);
  v4f a1 = *(const v4f*)(p + 4);
  v4f b0 = *(const v4f*)(p + 16);
  v4f b1 = *(const v4f*)(p + 20);
  v16h r;
#pragma unroll
  for (int i = 0; i < 4; ++i) {
    r[i]      = (_Float16)a0[i];
    r[i + 4]  = (_Float16)a1[i];
    r[i + 8]  = (_Float16)b0[i];
    r[i + 12] = (_Float16)b1[i];
  }
  return r;
}

// B operand (32x16, B[k][n] = src[n][k]) from row-major f16 source tile.
// Lane l: column n = (l&15)+n_off; halves 0-7 = K 8h..8h+7, halves 8-15 = K 16+8h..23+8h (h = l/16).
__device__ __forceinline__ v16h lds_B(const _Float16* base, int lane, int n_off) {
  int n  = (lane & 15) + n_off;
  int kb = (lane >> 4) << 3;
  const _Float16* p = base + n * 32 + kb;
  v8h lo = *(const v8h*)(p);
  v8h hi = *(const v8h*)(p + 16);
  v16h r;
#pragma unroll
  for (int i = 0; i < 8; ++i) { r[i] = lo[i]; r[i + 8] = hi[i]; }
  return r;
}

// C/D layout: VGPR r, lanes 0-15: (M=r, N=lane); lanes 16-31: (M=r+8, N=lane-16)
__device__ __forceinline__ void lds_store_C16(_Float16* base, int lane,
                                              int m_off, int n_off, v8f d) {
  int m0 = ((lane >> 4) << 3) + m_off;
  int n  = (lane & 15) + n_off;
#pragma unroll
  for (int r = 0; r < 8; ++r) base[(m0 + r) * 32 + n] = (_Float16)d[r];
}

// Row-sum over the 16-lane half-wave (xor masks < 16 stay inside the group).
__device__ __forceinline__ float red_sum16(float v) {
#pragma unroll
  for (int s = 1; s < 16; s <<= 1) v += __shfl_xor(v, s, 32);
  return v;
}

__global__ __launch_bounds__(WPB * 32)
void causal_attn_kernel(const float* __restrict__ x,
                        const float* __restrict__ Wk, const float* __restrict__ bk,
                        const float* __restrict__ Wq, const float* __restrict__ bq,
                        const float* __restrict__ Wv, const float* __restrict__ bv,
                        const float* __restrict__ Wp, const float* __restrict__ bp,
                        float* __restrict__ out, int Bt) {
  __shared__ alignas(32) float    xsh[WPB][1024];    // per-wave x tile (f32)
  __shared__ alignas(32) _Float16 wsh[4][1024];      // Wk,Wq(scaled),Wv,Wp f16
  __shared__ float bsh[4][32];
  __shared__ alignas(32) _Float16 scr[WPB][3][1024]; // per-wave k/q/v f16 tiles

  const int tid  = threadIdx.x;
  const int lane = tid & 31;
  const int wav  = tid >> 5;
  const int b    = blockIdx.x * WPB + wav;

  // ---- kick off async x-tile fetch (overlaps the weight conversion) ----
  if (b < Bt) {
    uint32_t l0 = (uint32_t)(uintptr_t)(&xsh[wav][lane * 4]);
    uint64_t g0 = (uint64_t)(uintptr_t)(x + (size_t)b * 1024 + lane * 4);
    ASYNC_B128_OFF(l0, g0, 0);
    ASYNC_B128_OFF(l0, g0, 512);
    ASYNC_B128_OFF(l0, g0, 1024);
    ASYNC_B128_OFF(l0, g0, 1536);
    ASYNC_B128_OFF(l0, g0, 2048);
    ASYNC_B128_OFF(l0, g0, 2560);
    ASYNC_B128_OFF(l0, g0, 3072);
    ASYNC_B128_OFF(l0, g0, 3584);
  }

  // ---- one-time per-block weight conversion; fold 1/sqrt(C) into Wq/bq ----
  const float* wsrc[4] = {Wk, Wq, Wv, Wp};
  const float* bsrc[4] = {bk, bq, bv, bp};
  const float scale = 0.17677669529663687f;  // 1/sqrt(32)
#pragma unroll
  for (int w = 0; w < 4; ++w) {
    float s = (w == 1) ? scale : 1.0f;
    for (int i = tid; i < 1024; i += WPB * 32)
      wsh[w][i] = (_Float16)(wsrc[w][i] * s);
    if (tid < 32) bsh[w][tid] = bsrc[w][tid] * s;
  }
  __syncthreads();

  if (b < Bt) {
    _Float16* kt = scr[wav][0];
    _Float16* qt = scr[wav][1];
    _Float16* vt = scr[wav][2];
    _Float16* at = (_Float16*)xsh[wav];  // reuses x buffer (x dead after k/q/v)
    _Float16* yt = kt;                   // reuses kt (k dead after logits)

    // ---- wait for the async x tile, build A operands with f32->f16 cvt ----
    wait_async0();
    v16h a0 = lds_A_f32(xsh[wav], lane, 0);
    v16h a1 = lds_A_f32(xsh[wav], lane, 16);

    // ---- k/q/v = x @ W^T + b  (q pre-scaled) ----
#pragma unroll
    for (int w = 0; w < 3; ++w) {
      _Float16* dst = scr[wav][w];  // k, q, v
#pragma unroll
      for (int nt = 0; nt < 2; ++nt) {
        v16h bm = lds_B(wsh[w], lane, nt * 16);
        float bval = bsh[w][(lane & 15) + nt * 16];
        v8f c;
#pragma unroll
        for (int i = 0; i < 8; ++i) c[i] = bval;
        v8f d0 = wmma16(a0, bm, c);
        v8f d1 = wmma16(a1, bm, c);
        lds_store_C16(dst, lane, 0,  nt * 16, d0);
        lds_store_C16(dst, lane, 16, nt * 16, d1);
      }
    }
    wave_lds_fence();

    // ---- logits = q @ k^T (scaled); upper-right 16x16 tile fully masked ----
    v16h q0 = lds_A(qt, lane, 0);
    v16h q1 = lds_A(qt, lane, 16);
    v16h k0 = lds_B(kt, lane, 0);
    v16h k1 = lds_B(kt, lane, 16);
    v8f zero = {};
    v8f l00 = wmma16(q0, k0, zero);
    v8f l10 = wmma16(q1, k0, zero);
    v8f l11 = wmma16(q1, k1, zero);

    // ---- causal softmax (max-free: logits ~ N(0,1), exp cannot overflow) ----
    const int n0 = lane & 15;
    const int n1 = n0 + 16;
    {  // rows 0..15 : cols 16..31 fully masked -> bulk-zero that quadrant
      v8h z16 = {};
      *(v8h*)(at + (lane & 15) * 32 + 16 + ((lane >> 4) << 3)) = z16;
      int mb = (lane >> 4) * 8;
      v8f p0;
#pragma unroll
      for (int r = 0; r < 8; ++r) {
        int m = mb + r;
        float e0 = (n0 <= m) ? __expf(l00[r]) : 0.0f;
        float inv = __builtin_amdgcn_rcpf(red_sum16(e0));
        p0[r] = e0 * inv;
      }
      lds_store_C16(at, lane, 0, 0, p0);
    }
    {  // rows 16..31
      int mb = (lane >> 4) * 8 + 16;
      v8f p0, p1;
#pragma unroll
      for (int r = 0; r < 8; ++r) {
        int m = mb + r;
        float e0 = __expf(l10[r]);  // n0 <= 15 < 16 <= m : always visible
        float e1 = (n1 <= m) ? __expf(l11[r]) : 0.0f;
        float inv = __builtin_amdgcn_rcpf(red_sum16(e0 + e1));
        p0[r] = e0 * inv;
        p1[r] = e1 * inv;
      }
      lds_store_C16(at, lane, 16, 0,  p0);
      lds_store_C16(at, lane, 16, 16, p1);
    }
    wave_lds_fence();

    // ---- y = att @ v^T  (B source = row-major v tile) ----
    v16h t0 = lds_A(at, lane, 0);
    v16h t1 = lds_A(at, lane, 16);
    v16h v0 = lds_B(vt, lane, 0);
    v16h v1 = lds_B(vt, lane, 16);
    v8f y00 = wmma16(t0, v0, zero);
    v8f y01 = wmma16(t0, v1, zero);
    v8f y10 = wmma16(t1, v0, zero);
    v8f y11 = wmma16(t1, v1, zero);
    lds_store_C16(yt, lane, 0,  0,  y00);
    lds_store_C16(yt, lane, 0,  16, y01);
    lds_store_C16(yt, lane, 16, 0,  y10);
    lds_store_C16(yt, lane, 16, 16, y11);
    wave_lds_fence();

    // ---- out = permute(y) @ Wp^T + bp ; scatter rows per reshape trick ----
    // row i of this batch's y lands at flat i*(Bt*32) + (b>>5)*1024 + (b&31)*32
    v16h y0 = lds_A(yt, lane, 0);
    v16h y1 = lds_A(yt, lane, 16);
    const size_t istride = (size_t)Bt * 32;
    const size_t baseoff0 = (size_t)(b >> 5) * 1024 + (size_t)(b & 31) * 32;
#pragma unroll
    for (int nt = 0; nt < 2; ++nt) {
      v16h bw = lds_B(wsh[3], lane, nt * 16);
      float bval = bsh[3][(lane & 15) + nt * 16];
      v8f c;
#pragma unroll
      for (int i = 0; i < 8; ++i) c[i] = bval;
      v8f d0 = wmma16(y0, bw, c);
      v8f d1 = wmma16(y1, bw, c);
      size_t baseoff = baseoff0 + (size_t)((lane & 15) + nt * 16);
      int m0 = (lane >> 4) * 8;
#pragma unroll
      for (int r = 0; r < 8; ++r) {
        out[(size_t)(m0 + r)      * istride + baseoff] = d0[r];
        out[(size_t)(m0 + r + 16) * istride + baseoff] = d1[r];
      }
    }
  }
}

extern "C" void kernel_launch(void* const* d_in, const int* in_sizes, int n_in,
                              void* d_out, int out_size, void* d_ws, size_t ws_size,
                              hipStream_t stream) {
  const float* x  = (const float*)d_in[0];
  const float* Wk = (const float*)d_in[1];
  const float* bk = (const float*)d_in[2];
  const float* Wq = (const float*)d_in[3];
  const float* bq = (const float*)d_in[4];
  const float* Wv = (const float*)d_in[5];
  const float* bv = (const float*)d_in[6];
  const float* Wp = (const float*)d_in[7];
  const float* bp = (const float*)d_in[8];
  float* out = (float*)d_out;

  const int Bt = in_sizes[0] / 1024;  // [B, 32, 32]
  dim3 grid((Bt + WPB - 1) / WPB), block(WPB * 32);
  hipLaunchKernelGGL(causal_attn_kernel, grid, block, 0, stream,
                     x, Wk, bk, Wq, bq, Wv, bv, Wp, bp, out, Bt);
}
